// SEMHSA_86620900426111
// MI455X (gfx1250) — matrix operand order repaired and checked by reading, not compiled, not run on hardware
//
#include <hip/hip_runtime.h>
#include <hip/hip_bf16.h>

typedef __attribute__((ext_vector_type(16))) _Float16 v16h;
typedef __attribute__((ext_vector_type(8)))  _Float16 v8h;
typedef __attribute__((ext_vector_type(8)))  float    v8f;
typedef unsigned int u32x4 __attribute__((ext_vector_type(4)));
typedef int          i32x8 __attribute__((ext_vector_type(8)));
typedef int          i32x4 __attribute__((ext_vector_type(4)));

#define NB    4
#define NC    192
#define NC3   576
#define NH    48
#define NTOK  2304          // 48*48
#define HEADS 6
#define HD    32
#define NG    24            // NB*HEADS

// ---------------------------------------------------------------------------
// Tensor Data Mover: build a D# (ISA ch.8) for a 2-D tile of 2-byte elements
// and kick an async global->LDS DMA.  Completion via TENSORcnt.
//   group0: [count=1 | lds_addr | global_addr(57b) | type=2]
//   group1: [data_size=2B | tensor_dim0/1 | tile_dim0/1 | dim0 stride]
// ---------------------------------------------------------------------------
__device__ __forceinline__ void tdm_load_2d(uint32_t lds_off, const void* gaddr,
                                            uint32_t dim0, uint32_t dim1,
                                            uint32_t tile0, uint32_t tile1,
                                            uint32_t stride0) {
    uint64_t ga = (uint64_t)(uintptr_t)gaddr;
    u32x4 g0;
    g0[0] = 1u;                                               // count = 1, user desc
    g0[1] = lds_off;                                          // LDS byte address
    g0[2] = (uint32_t)ga;                                     // global_addr[31:0]
    g0[3] = ((uint32_t)(ga >> 32) & 0x01FFFFFFu) | (2u << 30); // addr[56:32] | type=2
    i32x8 g1;
    g1[0] = (int)(1u << 16);                                  // data_size = 2 bytes
    g1[1] = (int)((dim0 & 0xFFFFu) << 16);                    // tensor_dim0[15:0]
    g1[2] = (int)(((dim0 >> 16) & 0xFFFFu) | ((dim1 & 0xFFFFu) << 16));
    g1[3] = (int)(((dim1 >> 16) & 0xFFFFu) | (tile0 << 16));  // tile_dim0
    g1[4] = (int)(tile1 & 0xFFFFu);                           // tile_dim1
    g1[5] = (int)stride0;                                     // tensor_dim0_stride
    g1[6] = 0;
    g1[7] = 0;
    i32x4 z4 = {};
#if defined(__clang_major__) && (__clang_major__ >= 23)
    i32x8 z8 = {};
    __builtin_amdgcn_tensor_load_to_lds(g0, g1, z4, z4, z8, 0);
#else
    __builtin_amdgcn_tensor_load_to_lds(g0, g1, z4, z4, 0);
#endif
}

// ---------------------------------------------------------------------------
// WMMA tile loaders (contiguous k = 16*(lane/16) + s in A and B, wave32)
// ---------------------------------------------------------------------------
__device__ __forceinline__ v16h load_A16x32(const _Float16* __restrict__ A, int lda,
                                            int m0, int k0, int lane) {
    int m  = m0 + (lane & 15);
    int kh = (lane >> 4) * 16;
    const _Float16* p = A + (long)m * lda + k0 + kh;
    v8h lo = *(const v8h*)(p);
    v8h hi = *(const v8h*)(p + 8);
    v16h a;
#pragma unroll
    for (int i = 0; i < 8; ++i) { a[i] = lo[i]; a[i + 8] = hi[i]; }
    return a;
}

__device__ __forceinline__ v16h load_B32x16(const _Float16* __restrict__ Bt, int ldb,
                                            int n0, int k0, int lane) {
    int n  = n0 + (lane & 15);
    int kb = (lane >> 4) * 16;
    return *(const v16h*)(Bt + (long)n * ldb + k0 + kb);
}

// ---------------------------------------------------------------------------
// WMMA GEMM with 2x2 register blocking: each wave computes a 32x32 C tile
// (4 WMMAs per k-step from 2 A-tiles + 2 B-tiles).  B given transposed.
// ---------------------------------------------------------------------------
__global__ void __launch_bounds__(256) gemm32_wmma(
        const _Float16* __restrict__ A, const _Float16* __restrict__ Bt,
        float* __restrict__ C, int M, int N, int K, long strideB, long strideC) {
    int lane = threadIdx.x & 31;
    int wave = threadIdx.x >> 5;
    int tiles_n = N >> 5;
    int wt = blockIdx.x * 8 + wave;
    if (wt >= (M >> 5) * tiles_n) return;
    int mt = wt / tiles_n, nt = wt % tiles_n;

    const _Float16* Bb = Bt + (long)blockIdx.z * strideB;
    float*          Cb = C  + (long)blockIdx.z * strideC;

    v8f acc00 = {}, acc01 = {}, acc10 = {}, acc11 = {};
    for (int k0 = 0; k0 < K; k0 += 32) {
        v16h a0 = load_A16x32(A,  K, mt * 32,      k0, lane);
        v16h a1 = load_A16x32(A,  K, mt * 32 + 16, k0, lane);
        v16h b0 = load_B32x16(Bb, K, nt * 32,      k0, lane);
        v16h b1 = load_B32x16(Bb, K, nt * 32 + 16, k0, lane);
        acc00 = __builtin_amdgcn_wmma_f32_16x16x32_f16(false, a0, false, b0, (short)0, acc00, false, false);
        acc01 = __builtin_amdgcn_wmma_f32_16x16x32_f16(false, a0, false, b1, (short)0, acc01, false, false);
        acc10 = __builtin_amdgcn_wmma_f32_16x16x32_f16(false, a1, false, b0, (short)0, acc10, false, false);
        acc11 = __builtin_amdgcn_wmma_f32_16x16x32_f16(false, a1, false, b1, (short)0, acc11, false, false);
    }
    int n    = nt * 32 + (lane & 15);
    int mrow = mt * 32 + (lane >> 4) * 8;
#pragma unroll
    for (int r = 0; r < 8; ++r) {
        Cb[(long)(mrow + r) * N + n]           = acc00[r];
        Cb[(long)(mrow + r) * N + n + 16]      = acc01[r];
        Cb[(long)(mrow + r + 16) * N + n]      = acc10[r];
        Cb[(long)(mrow + r + 16) * N + n + 16] = acc11[r];
    }
}

// ---------------------------------------------------------------------------
__global__ void cvt_f16(const float* __restrict__ src, _Float16* __restrict__ dst, int n) {
    int i = blockIdx.x * 256 + threadIdx.x;
    if (i < n) dst[i] = (_Float16)src[i];
}

// x[b,c,p] f32 -> xT[b,p,c] f16 (LDS tile transpose); grid (144,12,4), block (16,16)
__global__ void transpose_cvt(const float* __restrict__ x, _Float16* __restrict__ xT) {
    __shared__ float tile[16][17];
    int p0 = blockIdx.x * 16, c0 = blockIdx.y * 16, b = blockIdx.z;
    int tx = threadIdx.x, ty = threadIdx.y;
    tile[ty][tx] = x[((long)(b * NC + c0 + ty)) * NTOK + p0 + tx];
    __syncthreads();
    xT[((long)(b * NTOK + p0 + ty)) * NC + c0 + tx] = (_Float16)tile[tx][ty];
}

// Depthwise 3x3, pad 1, groups = 576
__global__ void dwconv3(const float* __restrict__ src, const float* __restrict__ wdw,
                        float* __restrict__ dst) {
    int idx = blockIdx.x * 256 + threadIdx.x;
    int p  = idx % NTOK;
    int ch = (idx / NTOK) % NC3;
    int b  = idx / (NTOK * NC3);
    int y = p / NH, x = p % NH;
    const float* s = src + ((long)(b * NC3 + ch)) * NTOK;
    const float* w = wdw + ch * 9;
    float acc = 0.f;
#pragma unroll
    for (int ky = 0; ky < 3; ++ky) {
        int yy = y + ky - 1;
        if (yy < 0 || yy >= NH) continue;
#pragma unroll
        for (int kx = 0; kx < 3; ++kx) {
            int xx = x + kx - 1;
            if (xx < 0 || xx >= NH) continue;
            acc += s[yy * NH + xx] * w[ky * 3 + kx];
        }
    }
    dst[idx] = acc;
}

// pdsca channel stats + SE MLP (one block per gi; gi<24 -> q, else k)
__global__ void pdsca_stats(const float* __restrict__ qkv,
                            const float* __restrict__ qw1, const float* __restrict__ qw2,
                            const float* __restrict__ kw1, const float* __restrict__ kw2,
                            float* __restrict__ meanmap, float* __restrict__ maxmap,
                            float* __restrict__ cabuf) {
    int gi  = blockIdx.x;
    int isK = gi >= NG;
    int g   = gi % NG;
    int b = g / HEADS, head = g % HEADS;
    int chbase = (isK ? NC : 0) + head * HD;
    const float* base = qkv + ((long)(b * NC3 + chbase)) * NTOK;

    __shared__ float csum[HD];
    __shared__ float hidden[8];
    if (threadIdx.x < HD) csum[threadIdx.x] = 0.f;
    __syncthreads();

    float local[HD];
#pragma unroll
    for (int d = 0; d < HD; ++d) local[d] = 0.f;

    for (int p = threadIdx.x; p < NTOK; p += 256) {
        float mean = 0.f, mx = -1e30f;
#pragma unroll
        for (int d = 0; d < HD; ++d) {
            float v = base[(long)d * NTOK + p];
            local[d] += v;
            mean += v;
            mx = fmaxf(mx, v);
        }
        meanmap[gi * NTOK + p] = mean * (1.f / HD);
        maxmap[gi * NTOK + p]  = mx;
    }
#pragma unroll
    for (int d = 0; d < HD; ++d) atomicAdd(&csum[d], local[d]);
    __syncthreads();

    const float* w1 = isK ? kw1 : qw1;   // [8,32]
    const float* w2 = isK ? kw2 : qw2;   // [32,8]
    if (threadIdx.x < 8) {
        float a = 0.f;
        for (int c = 0; c < HD; ++c) a += w1[threadIdx.x * HD + c] * (csum[c] * (1.f / NTOK));
        hidden[threadIdx.x] = fmaxf(a, 0.f);
    }
    __syncthreads();
    if (threadIdx.x < HD) {
        float a = 0.f;
        for (int r = 0; r < 8; ++r) a += w2[threadIdx.x * 8 + r] * hidden[r];
        cabuf[gi * HD + threadIdx.x] = 1.f / (1.f + __expf(-a));
    }
}

// 7x7 spatial conv over [mean,max], pad 3 -> sigmoid
__global__ void spatial_conv(const float* __restrict__ meanmap, const float* __restrict__ maxmap,
                             const float* __restrict__ qws, const float* __restrict__ kws,
                             float* __restrict__ sabuf) {
    int idx = blockIdx.x * 256 + threadIdx.x;
    int p = idx % NTOK, gi = idx / NTOK;
    int y = p / NH, x = p % NH;
    const float* ws = (gi < NG) ? qws : kws;
    const float* mm = meanmap + gi * NTOK;
    const float* xm = maxmap  + gi * NTOK;
    float acc = 0.f;
    for (int ky = 0; ky < 7; ++ky) {
        int yy = y + ky - 3;
        if (yy < 0 || yy >= NH) continue;
        for (int kx = 0; kx < 7; ++kx) {
            int xx = x + kx - 3;
            if (xx < 0 || xx >= NH) continue;
            int q = yy * NH + xx;
            acc += mm[q] * ws[ky * 7 + kx] + xm[q] * ws[49 + ky * 7 + kx];
        }
    }
    sabuf[idx] = 1.f / (1.f + __expf(-acc));
}

// tokens: q/k = l2norm(val * sigmoid(val*ca*sa)) -> [g,p,d] f16 ; v -> [g,d,p] f16
__global__ void build_tokens(const float* __restrict__ qkv,
                             const float* __restrict__ cabuf, const float* __restrict__ sabuf,
                             _Float16* __restrict__ qtok, _Float16* __restrict__ ktok,
                             _Float16* __restrict__ vT) {
    int idx = blockIdx.x * 256 + threadIdx.x;
    int p = idx % NTOK, g = idx / NTOK;
    int b = g / HEADS, head = g % HEADS;

#pragma unroll
    for (int which = 0; which < 2; ++which) {
        int gi = which * NG + g;
        const float* base = qkv + ((long)(b * NC3 + which * NC + head * HD)) * NTOK;
        const float* ca = cabuf + gi * HD;
        float sa = sabuf[gi * NTOK + p];
        float tmp[HD];
        float ssq = 0.f;
#pragma unroll
        for (int d = 0; d < HD; ++d) {
            float v = base[(long)d * NTOK + p];
            float mask = v * ca[d] * sa;
            float s = v * (1.f / (1.f + __expf(-mask)));
            tmp[d] = s;
            ssq += s * s;
        }
        float inv = 1.f / fmaxf(sqrtf(ssq), 1e-12f);
        _Float16* out = (which ? ktok : qtok) + (long)idx * HD;
#pragma unroll
        for (int d = 0; d < HD; ++d) out[d] = (_Float16)(tmp[d] * inv);
    }
    const float* vb = qkv + ((long)(b * NC3 + 2 * NC + head * HD)) * NTOK;
#pragma unroll
    for (int d = 0; d < HD; ++d)
        vT[((long)g * HD + d) * NTOK + p] = (_Float16)vb[(long)d * NTOK + p];
}

// ---------------------------------------------------------------------------
// Flash attention v3 (double-buffered TDM):
//  - 1 workgroup (8 waves) = one (b,head) x 256 queries; each wave owns 2
//    16-query tiles.
//  - K/V 32-key tiles are DMA'd into LDS by the Tensor Data Mover.  Two
//    staging buffers: while all 8 waves compute on tile i, the TDM streams
//    tile i+1.  TENSORcnt is in-order, so after issuing the next pair,
//    s_wait_tensorcnt(2) guarantees the current pair has landed.
//  - S-tile = 1 WMMA (d = 32 = K); online softmax with xor-shuffle row
//    reductions; P re-enters A layout via a per-wave LDS tile fenced with
//    s_wait_dscnt.
// ---------------------------------------------------------------------------
__global__ void __launch_bounds__(256) flash_attn(
        const _Float16* __restrict__ qtok, const _Float16* __restrict__ ktok,
        const _Float16* __restrict__ vT, _Float16* __restrict__ Obpc,
        const float* __restrict__ temp) {
    __shared__ __align__(128) char smem[2 * 4096 + 8 * 2048];  // 2 stage bufs + P bufs
    int lane = threadIdx.x & 31;
    int wave = threadIdx.x >> 5;
    _Float16* pbuf = (_Float16*)(smem + 8192) + wave * 1024;   // 2 P-tiles/wave

    int g     = blockIdx.x / (NTOK / 256);
    int qbase = (blockIdx.x % (NTOK / 256)) * 256 + wave * 32;
    int b = g / HEADS, head = g % HEADS;
    float scale = __expf(temp[0]);
    int m = lane & 15, half = lane >> 4;

    // two resident A_q tiles per wave
    v16h aq[2];
#pragma unroll
    for (int t = 0; t < 2; ++t) {
        const _Float16* qp = qtok + ((long)(g * NTOK + qbase + t * 16 + m) * HD) + half * 16;
        v8h lo = *(const v8h*)(qp);
        v8h hi = *(const v8h*)(qp + 8);
#pragma unroll
        for (int i = 0; i < 8; ++i) { aq[t][i] = lo[i]; aq[t][i + 8] = hi[i]; }
    }

    v8f acc[2][2] = {};
    float mrow[2][8], lrow[2][8];
#pragma unroll
    for (int t = 0; t < 2; ++t)
#pragma unroll
        for (int r = 0; r < 8; ++r) { mrow[t][r] = -1e30f; lrow[t][r] = 0.f; }

    const _Float16* kg = ktok + (long)g * NTOK * HD;
    const _Float16* vg = vT   + (long)g * HD * NTOK;
    const int NIT = NTOK / 32;

    // prologue: DMA tile 0 into stage buffer 0
    if (wave == 0) {
        tdm_load_2d(0,    kg, 1024, 1, 1024, 1, 1024);       // K: contiguous 2KB
        tdm_load_2d(2048, vg, HD, HD, HD, HD, NTOK);         // V: 32 rows, pitch NTOK
    }

    for (int it = 0; it < NIT; ++it) {
        int n0  = it * 32;
        int cur = (it & 1) * 4096;
        int nxt = ((it + 1) & 1) * 4096;
        if (wave == 0) {
            if (it + 1 < NIT) {
                // stream tile it+1 while tile it is consumed below
                tdm_load_2d(nxt,        kg + (long)(n0 + 32) * HD, 1024, 1, 1024, 1, 1024);
                tdm_load_2d(nxt + 2048, vg + n0 + 32,              HD, HD, HD, HD, NTOK);
                __builtin_amdgcn_s_wait_tensorcnt(2);  // in-order: pair(it) has landed
            } else {
                __builtin_amdgcn_s_wait_tensorcnt(0);
            }
        }
        __syncthreads();           // publish tile it to all 8 waves

        const _Float16* ktile = (const _Float16*)(smem + cur);
        const _Float16* vtile = (const _Float16*)(smem + cur + 2048);
        v16h bk0 = *(const v16h*)(ktile + m * HD + half * 16);
        v16h bk1 = *(const v16h*)(ktile + (16 + m) * HD + half * 16);
        v16h bv0 = *(const v16h*)(vtile + m * 32 + half * 16);
        v16h bv1 = *(const v16h*)(vtile + (16 + m) * 32 + half * 16);

#pragma unroll
        for (int t = 0; t < 2; ++t) {
            v8f z = {};
            v8f s0 = __builtin_amdgcn_wmma_f32_16x16x32_f16(false, aq[t], false, bk0, (short)0, z, false, false);
            v8f s1 = __builtin_amdgcn_wmma_f32_16x16x32_f16(false, aq[t], false, bk1, (short)0, z, false, false);
#pragma unroll
            for (int r = 0; r < 8; ++r) {
                float e0 = s0[r] * scale, e1 = s1[r] * scale;
                float mx = fmaxf(e0, e1);
                mx = fmaxf(mx, __shfl_xor(mx, 1, 32));
                mx = fmaxf(mx, __shfl_xor(mx, 2, 32));
                mx = fmaxf(mx, __shfl_xor(mx, 4, 32));
                mx = fmaxf(mx, __shfl_xor(mx, 8, 32));
                float mn = fmaxf(mrow[t][r], mx);
                float alpha = __expf(mrow[t][r] - mn);
                mrow[t][r] = mn;
                float p0 = __expf(e0 - mn), p1 = __expf(e1 - mn);
                float rs = p0 + p1;
                rs += __shfl_xor(rs, 1, 32);
                rs += __shfl_xor(rs, 2, 32);
                rs += __shfl_xor(rs, 4, 32);
                rs += __shfl_xor(rs, 8, 32);
                lrow[t][r] = lrow[t][r] * alpha + rs;
                acc[t][0][r] *= alpha;
                acc[t][1][r] *= alpha;
                pbuf[t * 512 + (r + half * 8) * 32 + m]      = (_Float16)p0;
                pbuf[t * 512 + (r + half * 8) * 32 + 16 + m] = (_Float16)p1;
            }
        }
        asm volatile("s_wait_dscnt 0x0" ::: "memory");

#pragma unroll
        for (int t = 0; t < 2; ++t) {
            const _Float16* pp = pbuf + t * 512 + m * 32 + half * 16;
            v8h plo = *(const v8h*)(pp);
            v8h phi = *(const v8h*)(pp + 8);
            v16h ap;
#pragma unroll
            for (int i = 0; i < 8; ++i) { ap[i] = plo[i]; ap[i + 8] = phi[i]; }
            acc[t][0] = __builtin_amdgcn_wmma_f32_16x16x32_f16(false, ap, false, bv0, (short)0, acc[t][0], false, false);
            acc[t][1] = __builtin_amdgcn_wmma_f32_16x16x32_f16(false, ap, false, bv1, (short)0, acc[t][1], false, false);
        }
        __syncthreads();   // all waves done with buffer `cur` before it is re-DMA'd
    }

#pragma unroll
    for (int t = 0; t < 2; ++t) {
        _Float16* ob = Obpc + ((long)(b * NTOK + qbase + t * 16) * NC) + head * HD;
#pragma unroll
        for (int r = 0; r < 8; ++r) {
            float inv = 1.f / lrow[t][r];
            int row = r + half * 8;
            ob[(long)row * NC + m]      = (_Float16)(acc[t][0][r] * inv);
            ob[(long)row * NC + 16 + m] = (_Float16)(acc[t][1][r] * inv);
        }
    }
}

// ---------------------------------------------------------------------------
extern "C" void kernel_launch(void* const* d_in, const int* in_sizes, int n_in,
                              void* d_out, int out_size, void* d_ws, size_t ws_size,
                              hipStream_t stream) {
    (void)in_sizes; (void)n_in; (void)out_size; (void)ws_size;
    const float* x     = (const float*)d_in[0];
    const float* w_qkv = (const float*)d_in[1];
    const float* w_dw  = (const float*)d_in[2];
    const float* q_w1  = (const float*)d_in[3];
    const float* q_w2  = (const float*)d_in[4];
    const float* q_ws  = (const float*)d_in[5];
    const float* k_w1  = (const float*)d_in[6];
    const float* k_w2  = (const float*)d_in[7];
    const float* k_ws  = (const float*)d_in[8];
    const float* w_prj = (const float*)d_in[9];
    const float* temp  = (const float*)d_in[10];
    float* out = (float*)d_out;

    char* ws = (char*)d_ws;
    size_t off = 0;
    auto alloc = [&](size_t bytes) {
        void* p = ws + off;
        off = (off + bytes + 255) & ~(size_t)255;
        return p;
    };
    _Float16* xT16    = (_Float16*)alloc((size_t)NB * NTOK * NC * 2);
    _Float16* wq16    = (_Float16*)alloc((size_t)NC3 * NC * 2);
    _Float16* wp16    = (_Float16*)alloc((size_t)NC * NC * 2);
    float*    qkv_pre = (float*)   alloc((size_t)NB * NC3 * NTOK * 4);
    float*    qkv     = (float*)   alloc((size_t)NB * NC3 * NTOK * 4);
    float*    meanmap = (float*)   alloc((size_t)2 * NG * NTOK * 4);
    float*    maxmap  = (float*)   alloc((size_t)2 * NG * NTOK * 4);
    float*    cabuf   = (float*)   alloc((size_t)2 * NG * HD * 4);
    float*    sabuf   = (float*)   alloc((size_t)2 * NG * NTOK * 4);
    _Float16* qtok    = (_Float16*)alloc((size_t)NG * NTOK * HD * 2);
    _Float16* ktok    = (_Float16*)alloc((size_t)NG * NTOK * HD * 2);
    _Float16* vT      = (_Float16*)alloc((size_t)NG * HD * NTOK * 2);
    _Float16* Obpc    = (_Float16*)alloc((size_t)NB * NTOK * NC * 2);

    cvt_f16<<<(NC3 * NC + 255) / 256, 256, 0, stream>>>(w_qkv, wq16, NC3 * NC);
    cvt_f16<<<(NC * NC + 255) / 256, 256, 0, stream>>>(w_prj, wp16, NC * NC);

    transpose_cvt<<<dim3(NTOK / 16, NC / 16, NB), dim3(16, 16), 0, stream>>>(x, xT16);

    // qkv = W_qkv * x : M=576, N=2304, K=192 (batched over b), 32x32 wave tiles
    gemm32_wmma<<<dim3((NC3 / 32) * (NTOK / 32) / 8, 1, NB), 256, 0, stream>>>(
        wq16, xT16, qkv_pre, NC3, NTOK, NC,
        (long)NTOK * NC, (long)NC3 * NTOK);

    dwconv3<<<(NB * NC3 * NTOK) / 256, 256, 0, stream>>>(qkv_pre, w_dw, qkv);

    pdsca_stats<<<2 * NG, 256, 0, stream>>>(qkv, q_w1, q_w2, k_w1, k_w2,
                                            meanmap, maxmap, cabuf);

    spatial_conv<<<(2 * NG * NTOK) / 256, 256, 0, stream>>>(meanmap, maxmap,
                                                            q_ws, k_ws, sabuf);

    build_tokens<<<(NG * NTOK) / 256, 256, 0, stream>>>(qkv, cabuf, sabuf,
                                                        qtok, ktok, vT);

    // flash attention: 24 heads x 9 query-blocks of 256
    flash_attn<<<NG * (NTOK / 256), 256, 0, stream>>>(qtok, ktok, vT, Obpc, temp);

    // projection: M=192, N=2304, K=192 (batched over b)
    gemm32_wmma<<<dim3((NC / 32) * (NTOK / 32) / 8, 1, NB), 256, 0, stream>>>(
        wp16, Obpc, out, NC, NTOK, NC,
        (long)NTOK * NC, (long)NC * NTOK);
}
